// GCN_9603546874307
// MI455X (gfx1250) — matrix operand order repaired and checked by reading, not compiled, not run on hardware
//
#include <hip/hip_runtime.h>
#include <hip/hip_bf16.h>
#include <math.h>

typedef __attribute__((ext_vector_type(16))) _Float16 v16h;
typedef __attribute__((ext_vector_type(8)))  float    v8f;
typedef __attribute__((ext_vector_type(4)))  float    f4;
typedef __attribute__((ext_vector_type(4)))  unsigned u32x4;
typedef __attribute__((ext_vector_type(8)))  int      i32x8;
typedef __attribute__((ext_vector_type(4)))  int      i32x4;

#if defined(__HIP_DEVICE_COMPILE__) && defined(__has_builtin)
# if __has_builtin(__builtin_amdgcn_tensor_load_to_lds) && __has_builtin(__builtin_amdgcn_s_wait_tensorcnt)
#  define USE_TDM 1
# endif
#endif
#ifndef USE_TDM
# define USE_TDM 0
#endif

// LDS row stride (f16 elements): 32 data + 8 pad = 80 bytes -> b128 frag reads
// are 16-byte aligned and 16 lanes map to 16 distinct bank groups (stride 20 banks).
#define LROW 40

// ---------------------------------------------------------------------------
// small utility kernels
// ---------------------------------------------------------------------------
__global__ void k_fill1(float* p, int n) {
    int i = blockIdx.x * blockDim.x + threadIdx.x;
    if (i < n) p[i] = 1.0f;                       // self-loop degree
}

__global__ void k_degree(const int* __restrict__ dst, float* __restrict__ deg, int E) {
    int i = blockIdx.x * blockDim.x + threadIdx.x;
    if (i < E) atomicAdd(&deg[dst[i]], 1.0f);
}

__global__ void k_rsqrt_inplace(float* p, int n) {
    int i = blockIdx.x * blockDim.x + threadIdx.x;
    if (i < n) p[i] = rsqrtf(p[i]);               // deg >= 1 always (self loops)
}

// W [K][Nw] f32  ->  Wt [Npad][K] f16 (transposed, zero padded rows)
__global__ void k_wt(const float* __restrict__ W, _Float16* __restrict__ Wt,
                     int K, int Nw, int Npad) {
    int i = blockIdx.x * blockDim.x + threadIdx.x;
    if (i >= Npad * K) return;
    int n = i / K, k = i - n * K;
    float v = (n < Nw) ? W[k * Nw + n] : 0.0f;
    Wt[i] = (_Float16)v;
}

// ---------------------------------------------------------------------------
// WMMA GEMM: C[M][ldc] = A[M][K](f32, optional relu) * Wt^T (Wt = [NT*16][K] f16)
// 256 threads = 8 waves; wave owns a 16-row slab across all NT column tiles.
// Double-buffered LDS; B tile staged by the Tensor Data Mover (wave 0), padding
// to the 80B LDS row stride done by TDM pad_interval/pad_amount.
// ---------------------------------------------------------------------------
template <int NT, bool RELU, bool NTA>
__global__ void k_gemm(const float* __restrict__ A, const _Float16* __restrict__ Wt,
                       float* __restrict__ C, int M, int K, int ldc) {
    __shared__ __align__(16) _Float16 As[2][128 * LROW];
    __shared__ __align__(16) _Float16 Bs[2][NT * 16 * LROW];

    const int tid  = threadIdx.x;
    const int lane = tid & 31;
    const int wave = tid >> 5;                    // 0..7
    const int row0 = blockIdx.x * 128;

    union Frag { v16h v; uint4 q[2]; };
    union CF   { v8f  v; float f[8]; };
    union Pack { uint2 u; _Float16 h[4]; };

    CF acc[NT];
    #pragma unroll
    for (int t = 0; t < NT; ++t)
        #pragma unroll
        for (int e = 0; e < 8; ++e) acc[t].f[e] = 0.0f;

    // --- stage A tile (128 x 32 f32 -> f16), vectorized float4 / uint2 ---
    auto stageA = [&](int k0, int buf) {
        const f4* Ag   = (const f4*)(A + (size_t)row0 * K + k0);
        const int ldf4 = K >> 2;
        if (row0 + 128 <= M) {
            #pragma unroll
            for (int j = 0; j < 4; ++j) {
                int q = tid + 256 * j;            // 1024 float4 per tile
                int r = q >> 3, c4 = q & 7;
                f4 v = NTA ? __builtin_nontemporal_load(&Ag[r * ldf4 + c4])
                           : Ag[r * ldf4 + c4];
                Pack p;
                #pragma unroll
                for (int e = 0; e < 4; ++e) {
                    float x = v[e];
                    if (RELU) x = fmaxf(x, 0.0f);
                    p.h[e] = (_Float16)x;
                }
                *(uint2*)&As[buf][r * LROW + c4 * 4] = p.u;
            }
        } else {
            #pragma unroll
            for (int j = 0; j < 4; ++j) {
                int q = tid + 256 * j;
                int r = q >> 3, c4 = q & 7;
                f4 v = {};
                if (row0 + r < M) v = Ag[r * ldf4 + c4];
                Pack p;
                #pragma unroll
                for (int e = 0; e < 4; ++e) {
                    float x = v[e];
                    if (RELU) x = fmaxf(x, 0.0f);
                    p.h[e] = (_Float16)x;
                }
                *(uint2*)&As[buf][r * LROW + c4 * 4] = p.u;
            }
        }
    };

    // --- stage B tile (NT*16 x 32 f16) ---
    auto stageB = [&](int k0, int buf) {
#if USE_TDM
        if (wave == 0) {
            // Tensor DMA: 2D tile, 2-byte elements, tile 32 x (NT*16),
            // LDS pad: after every 16 DWORDs (64B row) insert 4 DWORDs (16B).
            unsigned lds = (unsigned)(uintptr_t)&Bs[buf][0];
            unsigned long long ga =
                (unsigned long long)(uintptr_t)Wt + (unsigned long long)k0 * 2u;
            u32x4 g0 = { 1u,                                   // count=1
                         lds,                                  // lds_addr
                         (unsigned)ga,                         // global_addr lo
                         (unsigned)((ga >> 32) & 0x1FFFFFFu) | (2u << 30) }; // hi + type=2
            i32x8 g1 = { (1 << 16) | (1 << 20) | (3 << 22) | (3 << 25),
                         // ^ data_size=2B, pad_enable, pad_interval=16dw, pad_amount=4dw
                         (K & 0xFFFF) << 16,                   // tensor_dim0 lo16
                         ((K >> 16) & 0xFFFF) | ((NT * 16) << 16), // td0 hi | td1 lo
                         (32 << 16),                           // td1 hi | tile_dim0=32
                         (NT * 16),                            // tile_dim1 (tile_dim2=0)
                         K,                                    // tensor_dim0_stride lo32
                         0, 0 };
            i32x4 z4 = { 0, 0, 0, 0 };
            i32x8 z8 = { 0, 0, 0, 0, 0, 0, 0, 0 };
            __builtin_amdgcn_tensor_load_to_lds(g0, g1, z4, z4, z8, 0);
        }
#else
        for (int i = tid; i < NT * 16 * 8; i += 256) {        // uint2 = 4 f16
            int n = i >> 3, c4 = i & 7;
            uint2 w = *(const uint2*)&Wt[n * K + k0 + c4 * 4];
            *(uint2*)&Bs[buf][n * LROW + c4 * 4] = w;
        }
#endif
    };

    auto waitB = [&]() {
#if USE_TDM
        if (wave == 0) __builtin_amdgcn_s_wait_tensorcnt(0);
#endif
    };

    // --- pipelined main loop ---
    const int nk = K >> 5;
    stageA(0, 0);
    stageB(0, 0);
    waitB();
    __syncthreads();

    for (int kt = 0; kt < nk; ++kt) {
        const int cur = kt & 1, nxt = cur ^ 1;
        if (kt + 1 < nk) { stageA((kt + 1) << 5, nxt); stageB((kt + 1) << 5, nxt); }

        // preload A fragment + all NT B fragments, then a dense WMMA chain
        Frag a;
        const int ar = wave * 16 + (lane & 15);
        const int kb = (lane >> 4) * 8;
        const _Float16* ap = &As[cur][ar * LROW + kb];
        a.q[0] = *(const uint4*)ap;
        a.q[1] = *(const uint4*)(ap + 16);

        Frag b[NT];
        #pragma unroll
        for (int t = 0; t < NT; ++t) {
            const _Float16* bp = &Bs[cur][(t * 16 + (lane & 15)) * LROW + (lane >> 4) * 8];
            b[t].q[0] = *(const uint4*)bp;
            b[t].q[1] = *(const uint4*)(bp + 16);
        }
        #pragma unroll
        for (int t = 0; t < NT; ++t)
            acc[t].v = __builtin_amdgcn_wmma_f32_16x16x32_f16(
                false, a.v, false, b[t].v, (short)0, acc[t].v, false, false);

        if (kt + 1 < nk) { waitB(); __syncthreads(); }
    }

    // store: C layout — N = lane%16, M = v + 8*(lane/16) within 16-row slab
    // (32-bit offsets: M*ldc < 2^31 here)
    const int rbase = row0 + wave * 16 + ((lane >> 4) * 8);
    #pragma unroll
    for (int t = 0; t < NT; ++t) {
        const int off0 = rbase * ldc + t * 16 + (lane & 15);
        #pragma unroll
        for (int v = 0; v < 8; ++v) {
            if (rbase + v < M) C[off0 + v * ldc] = acc[t].f[v];
        }
    }
}

// ---------------------------------------------------------------------------
// aggregation: out[i] = t[i]*dinv[i]^2 + b   (self-loop + bias init)
// ---------------------------------------------------------------------------
__global__ void k_agg_init(const float* __restrict__ t, const float* __restrict__ dinv,
                           const float* __restrict__ b, float* __restrict__ out,
                           int Nn, int d, int ldt, int ldo) {
    int i = blockIdx.x * blockDim.x + threadIdx.x;   // Nn*d < 2^31
    if (i >= Nn * d) return;
    int row = i / d, f = i - row * d;
    float di = dinv[row];
    out[row * ldo + f] = t[row * ldt + f] * di * di + b[f];
}

// one wave per edge: out[dst] += t[src] * dinv[src]*dinv[dst]
__global__ void k_agg_edges(const float* __restrict__ t, const int* __restrict__ src,
                            const int* __restrict__ dst, const float* __restrict__ dinv,
                            float* __restrict__ out, int E, int d, int ldt, int ldo) {
    int wave = (blockIdx.x * blockDim.x + threadIdx.x) >> 5;
    int lane = threadIdx.x & 31;
    if (wave >= E) return;
    int s = src[wave], tg = dst[wave];
    float norm = dinv[s] * dinv[tg];
    int sb = s * ldt, tb = tg * ldo;                 // 32-bit offsets
    for (int f = lane; f < d; f += 32)
        atomicAdd(&out[tb + f], t[sb + f] * norm);
}

// one wave per row log_softmax over d (=40) classes
__global__ void k_logsoftmax(const float* __restrict__ logits, float* __restrict__ out,
                             int Nn, int d) {
    int row  = (blockIdx.x * blockDim.x + threadIdx.x) >> 5;
    int lane = threadIdx.x & 31;
    if (row >= Nn) return;
    const float* r = logits + row * d;
    float m = -INFINITY;
    for (int f = lane; f < d; f += 32) m = fmaxf(m, r[f]);
    for (int o = 16; o; o >>= 1) m = fmaxf(m, __shfl_xor(m, o, 32));
    float s = 0.0f;
    for (int f = lane; f < d; f += 32) s += expf(r[f] - m);
    for (int o = 16; o; o >>= 1) s += __shfl_xor(s, o, 32);
    float lse = m + logf(s);
    for (int f = lane; f < d; f += 32) out[row * d + f] = r[f] - lse;
}

// ---------------------------------------------------------------------------
static inline size_t al256(size_t x) { return (x + 255) & ~(size_t)255; }

extern "C" void kernel_launch(void* const* d_in, const int* in_sizes, int n_in,
                              void* d_out, int out_size, void* d_ws, size_t ws_size,
                              hipStream_t stream) {
    const float* x  = (const float*)d_in[0];
    const int*   ei = (const int*)d_in[1];
    const float* W1 = (const float*)d_in[2];
    const float* b1 = (const float*)d_in[3];
    const float* W2 = (const float*)d_in[4];
    const float* b2 = (const float*)d_in[5];

    const int dh   = in_sizes[3];              // 96
    const int dout = in_sizes[5];              // 40
    const int din  = in_sizes[2] / dh;         // 512
    const int Nn   = in_sizes[0] / din;        // 50000
    const int E    = in_sizes[1] / 2;          // 800000
    const int* src = ei;
    const int* dst = ei + E;

    const int NP1 = 96;                        // == dh (6 WMMA col tiles)
    const int NP2 = 48;                        // dout padded to 3 col tiles

    // workspace layout
    char* ws = (char*)d_ws;
    size_t o = 0;
    float*     dinv = (float*)(ws + o);     o += al256((size_t)Nn * 4);
    _Float16*  W1t  = (_Float16*)(ws + o);  o += al256((size_t)NP1 * din * 2);
    _Float16*  W2t  = (_Float16*)(ws + o);  o += al256((size_t)NP2 * dh * 2);
    float*     t1   = (float*)(ws + o);     o += al256((size_t)Nn * NP1 * 4);
    float*     h    = (float*)(ws + o);     o += al256((size_t)Nn * NP1 * 4);
    float*     t2   = (float*)(ws + o);     o += al256((size_t)Nn * NP2 * 4);

    float* logits = (float*)d_out;
    float* lsm    = logits + (size_t)Nn * dout;

    // degree -> dinv
    k_fill1<<<(Nn + 255) / 256, 256, 0, stream>>>(dinv, Nn);
    k_degree<<<(E + 255) / 256, 256, 0, stream>>>(dst, dinv, E);
    k_rsqrt_inplace<<<(Nn + 255) / 256, 256, 0, stream>>>(dinv, Nn);

    // weight transpose + f16 convert
    k_wt<<<(NP1 * din + 255) / 256, 256, 0, stream>>>(W1, W1t, din, dh, NP1);
    k_wt<<<(NP2 * dh + 255) / 256, 256, 0, stream>>>(W2, W2t, dh, dout, NP2);

    // layer 1: t1 = x @ W1 (WMMA, non-temporal x reads), aggregate + bias
    k_gemm<6, false, true><<<(Nn + 127) / 128, 256, 0, stream>>>(x, W1t, t1, Nn, din, NP1);
    k_agg_init<<<(Nn * dh + 255) / 256, 256, 0, stream>>>(t1, dinv, b1, h, Nn, dh, NP1, NP1);
    k_agg_edges<<<(E + 7) / 8, 256, 0, stream>>>(t1, src, dst, dinv, h, E, dh, NP1, NP1);

    // layer 2: t2 = relu(h) @ W2 (WMMA, relu fused into A load)
    k_gemm<3, true, false><<<(Nn + 127) / 128, 256, 0, stream>>>(h, W2t, t2, Nn, dh, NP2);
    k_agg_init<<<(Nn * dout + 255) / 256, 256, 0, stream>>>(t2, dinv, b2, logits, Nn, dout, NP2, dout);
    k_agg_edges<<<(E + 7) / 8, 256, 0, stream>>>(t2, src, dst, dinv, logits, E, dout, NP2, dout);

    // log_softmax into second half of d_out
    k_logsoftmax<<<(Nn + 7) / 8, 256, 0, stream>>>(logits, lsm, Nn, dout);
}
